// NNForceXYZ_29119878267395
// MI455X (gfx1250) — compile-verified, matrix operand order checked by reading, not run on hardware
//
#include <hip/hip_runtime.h>
#include <hip/hip_bf16.h>
#include <math.h>

typedef __attribute__((ext_vector_type(16))) _Float16 v16h;
typedef __attribute__((ext_vector_type(8)))  float    v8f;

#define KBT_CONST (0.00831446261815324f * 300.0f)
#define CHUNK 4096

struct V3 { float x, y, z; };
__device__ __forceinline__ V3 ld3(const float* __restrict__ p, int i) {
    return { p[3*i], p[3*i+1], p[3*i+2] };
}
__device__ __forceinline__ V3 sub3(V3 a, V3 b) { return { a.x-b.x, a.y-b.y, a.z-b.z }; }
__device__ __forceinline__ float dot3(V3 a, V3 b) { return a.x*b.x + a.y*b.y + a.z*b.z; }
__device__ __forceinline__ V3 cross3(V3 a, V3 b) {
    return { a.y*b.z - a.z*b.y, a.z*b.x - a.x*b.z, a.x*b.y - a.y*b.x };
}

// ---------------------------------------------------------------------------
// Kernel 1: geometry features. x = [cos(angles) | cos(dih) | sin(dih)]
// cos(atan2(y,x)) = x*rsqrt(x^2+y^2); sin likewise -> no trig needed.
// ---------------------------------------------------------------------------
__global__ void feat_kernel(const float* __restrict__ pos,
                            const int*   __restrict__ aidx,
                            const int*   __restrict__ didx,
                            float*       __restrict__ x,
                            int A, int D)
{
    int t = blockIdx.x * blockDim.x + threadIdx.x;
    if (t < A) {
        const int* q = aidx + 3*t;
        V3 p0 = ld3(pos, q[0]), p1 = ld3(pos, q[1]), p2 = ld3(pos, q[2]);
        V3 v0 = sub3(p0, p1);
        V3 v1 = sub3(p2, p1);
        float d  = dot3(v0, v1);
        V3 c  = cross3(v0, v1);
        float c2 = dot3(c, c);
        // angle = atan2(|c|, d); cos(angle) = d / sqrt(d^2 + |c|^2)
        x[t] = d * rsqrtf(d*d + c2);
    } else if (t < A + D) {
        int j = t - A;
        const int* q = didx + 4*j;
        V3 p0 = ld3(pos, q[0]), p1 = ld3(pos, q[1]);
        V3 p2 = ld3(pos, q[2]), p3 = ld3(pos, q[3]);
        V3 b1 = sub3(p1, p0);
        V3 b2 = sub3(p2, p1);
        V3 b3 = sub3(p3, p2);
        V3 n1 = cross3(b1, b2);
        V3 n2 = cross3(b2, b3);
        V3 m  = cross3(n1, n2);
        float yy = dot3(m, b2) * rsqrtf(dot3(b2, b2)); // (n1 x n2) . b2_unit
        float xx = dot3(n1, n2);
        float rinv = rsqrtf(xx*xx + yy*yy);
        x[A + j]     = xx * rinv;  // cos(dihedral)
        x[A + D + j] = yy * rinv;  // sin(dihedral)
    }
}

// ---------------------------------------------------------------------------
// Kernel 2: the 307 MB bandwidth-bound GEMV  part[row][chunk] = W1[row, ck].x[ck]
// float4 streaming loads; deterministic fixed-slot partial outputs.
// ---------------------------------------------------------------------------
__global__ void __launch_bounds__(256)
gemv_w1(const float* __restrict__ W1, const float* __restrict__ x,
        float* __restrict__ part, int F, int nch)
{
    int row = blockIdx.y;
    int ch  = blockIdx.x;
    int t   = threadIdx.x;
    const float* wr = W1 + (size_t)row * (size_t)F;
    int base = ch * CHUNK;

    float s = 0.0f;
#pragma unroll
    for (int i = 0; i < CHUNK / (256 * 4); ++i) {   // 4 iterations of float4
        int k = base + (i * 256 + t) * 4;
        if (k < F) {                                // F % 4 == 0, so whole-float4 guard
            float4 w  = *(const float4*)(wr + k);
            float4 xv = *(const float4*)(x + k);
            s = fmaf(w.x, xv.x, s);
            s = fmaf(w.y, xv.y, s);
            s = fmaf(w.z, xv.z, s);
            s = fmaf(w.w, xv.w, s);
        }
    }

    __shared__ float red[256];
    red[t] = s;
    __syncthreads();
#pragma unroll
    for (int off = 128; off > 0; off >>= 1) {
        if (t < off) red[t] += red[t + off];
        __syncthreads();
    }
    if (t == 0) part[row * nch + ch] = red[0];
}

// ---------------------------------------------------------------------------
// Kernel 3: MLP tail. 1 block, 512 threads = 16 waves (wave32).
// Layers W2/W3 (256x256) done with v_wmma_f32_16x16x32_f16: wave w owns rows
// [16w, 16w+16); B holds the activation K-chunk replicated across all 16
// columns, so every column of D is a valid copy of the 16 row results.
// ISA 16-bit A layout: lanes 0-15 -> K {0..7,16..23}, lanes 16-31 -> K {8..15,24..31}.
// ---------------------------------------------------------------------------
__device__ __forceinline__ v8f wave_layer256(const float* __restrict__ W,
                                             const _Float16* __restrict__ yin,
                                             int m, int half)
{
    v8f acc = {};
#pragma unroll
    for (int kb = 0; kb < 8; ++kb) {
        v16h a, b;
        const float*    wr = W   + m * 256 + kb * 32;
        const _Float16* yk = yin + kb * 32;
#pragma unroll
        for (int v = 0; v < 8; ++v) {
            int kBase = (v < 4) ? (half * 8 + 2 * v)
                                : (16 + half * 8 + 2 * (v - 4));
            a[2*v]   = (_Float16)wr[kBase];
            a[2*v+1] = (_Float16)wr[kBase + 1];
            b[2*v]   = yk[kBase];
            b[2*v+1] = yk[kBase + 1];
        }
        acc = __builtin_amdgcn_wmma_f32_16x16x32_f16(
                  false, a, false, b, (short)0, acc, false, false);
    }
    return acc;
}

__global__ void __launch_bounds__(512)
mlp_tail(const float* __restrict__ part, int nch,
         const float* __restrict__ b1,
         const float* __restrict__ W2, const float* __restrict__ b2,
         const float* __restrict__ W3, const float* __restrict__ b3,
         const float* __restrict__ W4,
         float* __restrict__ out)
{
    __shared__ _Float16 hA[256];   // y1 (f16 input to W2)
    __shared__ _Float16 hB[256];   // h2 (f16 input to W3)
    __shared__ float    hF[256];   // h3 (f32 for final dot)
    __shared__ float    red[256];

    int t = threadIdx.x;

    // Stage 0: y1 = tanh(b1 + sum of W1 partials), fixed summation order.
    if (t < 256) {
        float s = b1[t];
        const float* p = part + t * nch;
        for (int c = 0; c < nch; ++c) s += p[c];
        hA[t] = (_Float16)tanhf(s);
    }
    __syncthreads();

    int lane    = t & 31;
    int rowBase = (t >> 5) * 16;
    int half    = lane >> 4;
    int m       = rowBase + (lane & 15);

    // Stage 1: h2 = tanh(W2 @ y1 + b2) via WMMA
    {
        v8f acc = wave_layer256(W2, hA, m, half);
        // D layout: VGPR v, lanes 0-15 -> M=v, lanes 16-31 -> M=v+8 (any column).
        if ((lane & 15) == 0) {
#pragma unroll
            for (int v = 0; v < 8; ++v) {
                int r = rowBase + half * 8 + v;
                hB[r] = (_Float16)tanhf(acc[v] + b2[r]);
            }
        }
    }
    __syncthreads();

    // Stage 2: h3 = tanh(W3 @ h2 + b3) via WMMA
    {
        v8f acc = wave_layer256(W3, hB, m, half);
        if ((lane & 15) == 0) {
#pragma unroll
            for (int v = 0; v < 8; ++v) {
                int r = rowBase + half * 8 + v;
                hF[r] = tanhf(acc[v] + b3[r]);
            }
        }
    }
    __syncthreads();

    // Stage 3: out = kB*T * (W4 . h3)
    if (t < 256) red[t] = W4[t] * hF[t];
    __syncthreads();
#pragma unroll
    for (int off = 128; off > 0; off >>= 1) {
        if (t < off) red[t] += red[t + off];
        __syncthreads();
    }
    if (t == 0) out[0] = KBT_CONST * red[0];
}

// ---------------------------------------------------------------------------
extern "C" void kernel_launch(void* const* d_in, const int* in_sizes, int n_in,
                              void* d_out, int out_size, void* d_ws, size_t ws_size,
                              hipStream_t stream)
{
    const float* pos  = (const float*)d_in[0];
    const int*   aidx = (const int*)  d_in[1];
    const int*   didx = (const int*)  d_in[2];
    const float* W1   = (const float*)d_in[3];
    const float* b1   = (const float*)d_in[4];
    const float* W2   = (const float*)d_in[5];
    const float* b2   = (const float*)d_in[6];
    const float* W3   = (const float*)d_in[7];
    const float* b3   = (const float*)d_in[8];
    const float* W4   = (const float*)d_in[9];

    int A = in_sizes[1] / 3;
    int D = in_sizes[2] / 4;
    int F = A + 2 * D;
    int nch = (F + CHUNK - 1) / CHUNK;

    float* x    = (float*)d_ws;   // F floats
    float* part = x + F;          // 256 * nch floats

    int nfeatThreads = A + D;
    feat_kernel<<<(nfeatThreads + 255) / 256, 256, 0, stream>>>(
        pos, aidx, didx, x, A, D);

    gemv_w1<<<dim3(nch, 256), 256, 0, stream>>>(W1, x, part, F, nch);

    mlp_tail<<<1, 512, 0, stream>>>(part, nch, b1, W2, b2, W3, b3, W4,
                                    (float*)d_out);
}
